// Attention_13254269075532
// MI455X (gfx1250) — matrix operand order repaired and checked by reading, not compiled, not run on hardware
//
#include <hip/hip_runtime.h>
#include <math.h>

typedef __attribute__((ext_vector_type(16))) _Float16 v16h;
typedef __attribute__((ext_vector_type(8)))  _Float16 v8h;
typedef __attribute__((ext_vector_type(8)))  float    v8f;

#define L_SEQ   2048
#define DIM     2048
#define NH      32
#define NKV     8
#define HD      64
#define KVD     (NKV*HD)     /* 512  */
#define QD      (NH*HD)      /* 2048 */
#define ATT_SCALE 0.125f     /* 64^-0.5 */

#define USE_ASYNC_COPY 1

// ---------------------------------------------------------------------------
// 16-halfword (32B) global->LDS copy via CDNA5 GLOBAL_LOAD_ASYNC_TO_LDS_B128
// (ASYNCcnt). LDS byte offset = low 32 bits of the flat LDS pointer.
// Each cp16 issues 2 async instructions per wave.
// ---------------------------------------------------------------------------
__device__ __forceinline__ void cp16(_Float16* lds, const _Float16* g) {
#if USE_ASYNC_COPY
  unsigned l0 = (unsigned)(size_t)(void*)lds;
  asm volatile("global_load_async_to_lds_b128 %0, %2, off\n\t"
               "global_load_async_to_lds_b128 %1, %3, off"
               :: "v"(l0), "v"(l0 + 16u), "v"(g), "v"(g + 8)
               : "memory");
#else
  *(v8h*)lds       = *(const v8h*)g;
  *(v8h*)(lds + 8) = *(const v8h*)(g + 8);
#endif
}
// drain everything
__device__ __forceinline__ void cp_wait0() {
#if USE_ASYNC_COPY
  asm volatile("s_wait_asynccnt 0x0" ::: "memory");
#endif
}
// allow the 4 newest ops (one prefetched double-buffer stage) to stay in flight
__device__ __forceinline__ void cp_wait4() {
#if USE_ASYNC_COPY
  asm volatile("s_wait_asynccnt 0x4" ::: "memory");
#endif
}

// ---- WMMA fragment loads (wave32, 16-bit data), 16B-aligned vector ops.
// A 16x32 (MxK): lane M = lane&15, hk = lane>>4: e<8 -> K = 8hk+e ; e>=8 -> K = 16+8hk+(e-8)
__device__ __forceinline__ v16h ld_frag_a(const _Float16* rowp, int hk) {
  v8h lo = *(const v8h*)(rowp + 8 * hk);
  v8h hi = *(const v8h*)(rowp + 16 + 8 * hk);
  return __builtin_shufflevector(lo, hi, 0,1,2,3,4,5,6,7,8,9,10,11,12,13,14,15);
}
// B 32x16 (KxN): lane N = lane&15, hk = lane>>4: e<8 -> K = 8hk+e ; e>=8 -> K = 16+8hk+(e-8)
__device__ __forceinline__ v16h ld_frag_b(const _Float16* rowp, int hk) {
  v8h lo = *(const v8h*)(rowp + 8 * hk);
  v8h hi = *(const v8h*)(rowp + 16 + 8 * hk);
  return __builtin_shufflevector(lo, hi, 0,1,2,3,4,5,6,7,8,9,10,11,12,13,14,15);
}

// ---------------------------------------------------------------------------
// fp32 -> f16 elementwise convert
// ---------------------------------------------------------------------------
__global__ void cvt_f16_kernel(const float* __restrict__ s, _Float16* __restrict__ d, int n) {
  int i = blockIdx.x * 256 + threadIdx.x;
  if (i < n) d[i] = (_Float16)s[i];
}

// ---------------------------------------------------------------------------
// fp32 [K,N] -> f16 [N,K] tiled transpose (coalesced both sides via LDS)
// ---------------------------------------------------------------------------
__global__ __launch_bounds__(256)
void transpose_cvt_kernel(const float* __restrict__ src, _Float16* __restrict__ dst,
                          int K, int N) {
  __shared__ _Float16 t[32][33];
  int nb = blockIdx.x * 32, kb = blockIdx.y * 32;
  int tx = threadIdx.x & 31, ty = threadIdx.x >> 5;
#pragma unroll
  for (int i = 0; i < 4; ++i) {
    int r = ty + 8 * i;
    t[r][tx] = (_Float16)src[(size_t)(kb + r) * N + nb + tx];
  }
  __syncthreads();
#pragma unroll
  for (int i = 0; i < 4; ++i) {
    int r = ty + 8 * i;
    dst[(size_t)(nb + r) * K + kb + tx] = t[tx][r];   // dst[n][k] = src[k][n]
  }
}

// ---------------------------------------------------------------------------
// TN GEMM, f16 inputs: C[M,N] = A[M,K] * BT[N,K]^T, fp32 out.
// Block 256 thr (8 waves), 64x64 C tile, K staged 64 at a time,
// double-buffered LDS with async prefetch of the next stage.
// ---------------------------------------------------------------------------
__global__ __launch_bounds__(256)
void gemm_tn_f16(const _Float16* __restrict__ A, const _Float16* __restrict__ BT,
                 float* __restrict__ C, int M, int N, int K) {
  __shared__ _Float16 As[2][64][72];
  __shared__ _Float16 Bs[2][64][72];

  const int tid  = threadIdx.x;
  const int lane = tid & 31, wave = tid >> 5;
  const int wr = wave >> 1, wc = wave & 1;
  const int n  = lane & 15, hk = lane >> 4;
  const int row0 = blockIdx.y * 64, col0 = blockIdx.x * 64;

  const int sr = tid >> 2, scc = (tid & 3) * 16;       // staging: 16 halves/thread
  const _Float16* ga = A  + (size_t)(row0 + sr) * K + scc;
  const _Float16* gb = BT + (size_t)(col0 + sr) * K + scc;

  v8f acc0, acc1;
#pragma unroll
  for (int i = 0; i < 8; ++i) { acc0[i] = 0.f; acc1[i] = 0.f; }

  cp16(&As[0][sr][scc], ga);               // stage 0
  cp16(&Bs[0][sr][scc], gb);

  int buf = 0;
  for (int k0 = 0; k0 < K; k0 += 64, buf ^= 1) {
    if (k0 + 64 < K) {                      // prefetch next stage, then wait
      cp16(&As[buf ^ 1][sr][scc], ga + k0 + 64);
      cp16(&Bs[buf ^ 1][sr][scc], gb + k0 + 64);
      cp_wait4();                           // current stage landed, next in flight
    } else {
      cp_wait0();
    }
    __syncthreads();
#pragma unroll
    for (int ks = 0; ks < 2; ++ks) {
      v16h a  = ld_frag_a(&As[buf][wr * 16 + n][ks * 32], hk);
      v16h b0 = ld_frag_b(&Bs[buf][wc * 32 + n][ks * 32], hk);
      v16h b1 = ld_frag_b(&Bs[buf][wc * 32 + 16 + n][ks * 32], hk);
      acc0 = __builtin_amdgcn_wmma_f32_16x16x32_f16(false, a, false, b0, (short)0, acc0, false, false);
      acc1 = __builtin_amdgcn_wmma_f32_16x16x32_f16(false, a, false, b1, (short)0, acc1, false, false);
    }
    __syncthreads();
  }

#pragma unroll
  for (int r = 0; r < 8; ++r) {                        // C: M = r + 8*hk, N = lane&15
    int row = row0 + wr * 16 + r + 8 * hk;
    C[(size_t)row * N + col0 + wc * 32 + n]      = acc0[r];
    C[(size_t)row * N + col0 + wc * 32 + 16 + n] = acc1[r];
  }
}

// ---------------------------------------------------------------------------
// RoPE in place on [L, nheads*64] fp32 (interleaved pairs == reference R)
// ---------------------------------------------------------------------------
__global__ void rope_kernel(float* __restrict__ buf, int nheads) {
  int idx = blockIdx.x * blockDim.x + threadIdx.x;
  int total = L_SEQ * nheads * (HD / 2);
  if (idx >= total) return;
  int m = idx % (HD / 2);
  int t = idx / (HD / 2);
  int h = t % nheads;
  int l = t / nheads;
  float theta = expf(-(float)m * (logf(10000.0f) / 32.0f));
  float ang = (float)l * theta;
  float s, c;
  sincosf(ang, &s, &c);
  float* p = buf + ((size_t)l * nheads + h) * HD + 2 * m;
  float x0 = p[0], x1 = p[1];
  p[0] = c * x0 - s * x1;
  p[1] = s * x0 + c * x1;
}

// ---------------------------------------------------------------------------
// Outputs 2 & 3: rotated K and V repeated to 32 heads ([B,H,L,d], fp32)
// ---------------------------------------------------------------------------
__global__ void kv_write_kernel(const float* __restrict__ Kr, const float* __restrict__ V,
                                float* __restrict__ outK, float* __restrict__ outV) {
  int idx = blockIdx.x * blockDim.x + threadIdx.x;
  if (idx >= NH * L_SEQ * HD) return;
  int d = idx & 63;
  int l = (idx >> 6) & (L_SEQ - 1);
  int h = idx >> 17;
  int kvh = h >> 2;
  size_t src = (size_t)l * KVD + kvh * HD + d;
  outK[idx] = Kr[src];
  outV[idx] = V[src];
}

// ---------------------------------------------------------------------------
// Fused causal flash attention, f16 inputs (Qh [L,2048], Kh [L,512],
// VT [512,2048] = V^T), writes AOh f16 [L,2048]. Grid (head, q-tile 64).
// K/V tiles double-buffered: next tile's async copy overlaps S-compute,
// softmax and P*V of the current tile.
// ---------------------------------------------------------------------------
__global__ __launch_bounds__(256)
void flash_attn_kernel(const _Float16* __restrict__ Qh, const _Float16* __restrict__ Kh,
                       const _Float16* __restrict__ VT, _Float16* __restrict__ AOh) {
  __shared__ _Float16 Qs[64][72];
  __shared__ _Float16 Ks[2][64][72];
  __shared__ _Float16 Vt[2][64][72];   // Vt[b][d][l_local]
  __shared__ _Float16 Ps[64][72];
  __shared__ float    Ss[64][65];
  __shared__ float    mS[64], lS[64], aS[64];

  const int tid  = threadIdx.x;
  const int lane = tid & 31, wave = tid >> 5;
  const int wr = wave >> 1, wc = wave & 1;
  const int n  = lane & 15, hk = lane >> 4;
  const int h  = blockIdx.x, qt = blockIdx.y;
  const int kvh = h >> 2;
  const int q0  = qt * 64;

  const int sr = tid >> 2, scc = (tid & 3) * 16;

  cp16(&Qs[sr][scc], Qh + (size_t)(q0 + sr) * QD + h * HD + scc);
  cp16(&Ks[0][sr][scc], Kh + (size_t)sr * KVD + kvh * HD + scc);            // jt = 0
  cp16(&Vt[0][sr][scc], VT + (size_t)(kvh * HD + sr) * L_SEQ + scc);
  if (tid < 64) { mS[tid] = -1.0e30f; lS[tid] = 0.f; }

  v8f o0, o1;
#pragma unroll
  for (int i = 0; i < 8; ++i) { o0[i] = 0.f; o1[i] = 0.f; }

  int buf = 0;
  for (int jt = 0; jt <= qt; ++jt, buf ^= 1) {
    int j0 = jt * 64;
    if (jt < qt) {                         // prefetch next key tile
      int jn = j0 + 64;
      cp16(&Ks[buf ^ 1][sr][scc], Kh + (size_t)(jn + sr) * KVD + kvh * HD + scc);
      cp16(&Vt[buf ^ 1][sr][scc], VT + (size_t)(kvh * HD + sr) * L_SEQ + jn + scc);
      cp_wait4();                          // current tile (and Qs) landed
    } else {
      cp_wait0();
    }
    __syncthreads();

    // S = Q * K^T : A = Qs rows (q), B[k][n] = Ks[n][k] -> rows of Ks
    v8f s0, s1;
#pragma unroll
    for (int i = 0; i < 8; ++i) { s0[i] = 0.f; s1[i] = 0.f; }
#pragma unroll
    for (int ks = 0; ks < 2; ++ks) {
      v16h a  = ld_frag_a(&Qs[wr * 16 + n][ks * 32], hk);
      v16h b0 = ld_frag_b(&Ks[buf][wc * 32 + n][ks * 32], hk);
      v16h b1 = ld_frag_b(&Ks[buf][wc * 32 + 16 + n][ks * 32], hk);
      s0 = __builtin_amdgcn_wmma_f32_16x16x32_f16(false, a, false, b0, (short)0, s0, false, false);
      s1 = __builtin_amdgcn_wmma_f32_16x16x32_f16(false, a, false, b1, (short)0, s1, false, false);
    }
    // scale + causal mask at fragment store
#pragma unroll
    for (int r = 0; r < 8; ++r) {
      int row = wr * 16 + r + 8 * hk;
      int c0 = wc * 32 + n, c1 = wc * 32 + 16 + n;
      float v0 = s0[r] * ATT_SCALE;
      float v1 = s1[r] * ATT_SCALE;
      if (q0 + row < j0 + c0) v0 -= 1.0e9f;
      if (q0 + row < j0 + c1) v1 -= 1.0e9f;
      Ss[row][c0] = v0;
      Ss[row][c1] = v1;
    }
    __syncthreads();

    // fp32 online softmax, one thread per row
    if (tid < 64) {
      int row = tid;
      float mold = mS[row];
      float mx = mold;
      for (int c = 0; c < 64; ++c) mx = fmaxf(mx, Ss[row][c]);
      float alpha = expf(mold - mx);
      float sum = 0.f;
      for (int c = 0; c < 64; ++c) {
        float p = expf(Ss[row][c] - mx);
        sum += p;
        Ps[row][c] = (_Float16)p;
      }
      lS[row] = lS[row] * alpha + sum;
      mS[row] = mx;
      aS[row] = alpha;
    }
    __syncthreads();

    // rescale O, then O += P * V : A = Ps rows, B[k][n] = Vt[n][k]
#pragma unroll
    for (int r = 0; r < 8; ++r) {
      float al = aS[wr * 16 + r + 8 * hk];
      o0[r] *= al;
      o1[r] *= al;
    }
#pragma unroll
    for (int ks = 0; ks < 2; ++ks) {
      v16h a  = ld_frag_a(&Ps[wr * 16 + n][ks * 32], hk);
      v16h b0 = ld_frag_b(&Vt[buf][wc * 32 + n][ks * 32], hk);
      v16h b1 = ld_frag_b(&Vt[buf][wc * 32 + 16 + n][ks * 32], hk);
      o0 = __builtin_amdgcn_wmma_f32_16x16x32_f16(false, a, false, b0, (short)0, o0, false, false);
      o1 = __builtin_amdgcn_wmma_f32_16x16x32_f16(false, a, false, b1, (short)0, o1, false, false);
    }
    __syncthreads();
  }

  // normalize and store f16 attention output [L, H*d]
#pragma unroll
  for (int r = 0; r < 8; ++r) {
    int row = wr * 16 + r + 8 * hk;
    float invl = 1.f / lS[row];
    size_t base = (size_t)(q0 + row) * QD + h * HD;
    AOh[base + wc * 32 + n]      = (_Float16)(o0[r] * invl);
    AOh[base + wc * 32 + 16 + n] = (_Float16)(o1[r] * invl);
  }
}

// ---------------------------------------------------------------------------
extern "C" void kernel_launch(void* const* d_in, const int* in_sizes, int n_in,
                              void* d_out, int out_size, void* d_ws, size_t ws_size,
                              hipStream_t stream) {
  (void)in_sizes; (void)n_in; (void)out_size; (void)ws_size;
  const float* x  = (const float*)d_in[0];
  // d_in[1]=rotation_matrix, d_in[2]=mask -> recomputed inline
  const float* wq = (const float*)d_in[3];
  const float* wk = (const float*)d_in[4];
  const float* wv = (const float*)d_in[5];
  const float* wo = (const float*)d_in[6];
  float* out = (float*)d_out;

  char* ws = (char*)d_ws;
  const size_t MB = 1024 * 1024;
  float*    Q   = (float*)(ws);                 // [2048,2048] fp32   16MB
  float*    K   = (float*)(ws + 16 * MB);       // [2048, 512] fp32    4MB
  float*    V   = (float*)(ws + 20 * MB);       // [2048, 512] fp32    4MB
  _Float16* xh  = (_Float16*)(ws + 24 * MB);    // [2048,2048] f16     8MB
  _Float16* wqT = (_Float16*)(ws + 32 * MB);    // [2048,2048] f16     8MB
  _Float16* wkT = (_Float16*)(ws + 40 * MB);    // [ 512,2048] f16     2MB
  _Float16* wvT = (_Float16*)(ws + 42 * MB);    // [ 512,2048] f16     2MB
  _Float16* woT = (_Float16*)(ws + 44 * MB);    // [2048,2048] f16     8MB
  _Float16* Qh  = (_Float16*)(ws + 52 * MB);    // [2048,2048] f16     8MB
  _Float16* Kh  = (_Float16*)(ws + 60 * MB);    // [2048, 512] f16     2MB
  _Float16* VT  = (_Float16*)(ws + 62 * MB);    // [ 512,2048] f16     2MB
  _Float16* AOh = (_Float16*)(ws + 64 * MB);    // [2048,2048] f16     8MB

  dim3 blk(256);

  // one-time f16 conversions / weight transposes (K-major everywhere)
  cvt_f16_kernel<<<(L_SEQ * DIM + 255) / 256, blk, 0, stream>>>(x, xh, L_SEQ * DIM);
  transpose_cvt_kernel<<<dim3(QD / 32, DIM / 32), blk, 0, stream>>>(wq, wqT, DIM, QD);
  transpose_cvt_kernel<<<dim3(KVD / 32, DIM / 32), blk, 0, stream>>>(wk, wkT, DIM, KVD);
  transpose_cvt_kernel<<<dim3(KVD / 32, DIM / 32), blk, 0, stream>>>(wv, wvT, DIM, KVD);
  transpose_cvt_kernel<<<dim3(DIM / 32, QD / 32), blk, 0, stream>>>(wo, woT, QD, DIM);

  // QKV projections (fp32 out for exact RoPE / kv outputs)
  gemm_tn_f16<<<dim3(QD / 64, L_SEQ / 64), blk, 0, stream>>>(xh, wqT, Q, L_SEQ, QD, DIM);
  gemm_tn_f16<<<dim3(KVD / 64, L_SEQ / 64), blk, 0, stream>>>(xh, wkT, K, L_SEQ, KVD, DIM);
  gemm_tn_f16<<<dim3(KVD / 64, L_SEQ / 64), blk, 0, stream>>>(xh, wvT, V, L_SEQ, KVD, DIM);

  // RoPE on Q (32 heads) and K (8 kv heads), fp32 in place
  int nq = L_SEQ * NH * (HD / 2);
  rope_kernel<<<(nq + 255) / 256, blk, 0, stream>>>(Q, NH);
  int nk = L_SEQ * NKV * (HD / 2);
  rope_kernel<<<(nk + 255) / 256, blk, 0, stream>>>(K, NKV);

  // outputs 2 & 3
  int nkv = NH * L_SEQ * HD;
  kv_write_kernel<<<(nkv + 255) / 256, blk, 0, stream>>>(K, V, out + nkv, out + 2 * (size_t)nkv);

  // f16 operands for attention: Qh, Kh row-major; V transposed
  cvt_f16_kernel<<<(L_SEQ * QD + 255) / 256, blk, 0, stream>>>(Q, Qh, L_SEQ * QD);
  cvt_f16_kernel<<<(L_SEQ * KVD + 255) / 256, blk, 0, stream>>>(K, Kh, L_SEQ * KVD);
  transpose_cvt_kernel<<<dim3(KVD / 32, L_SEQ / 32), blk, 0, stream>>>(V, VT, L_SEQ, KVD);

  // fused causal GQA attention -> AOh f16 [L, H*d]
  flash_attn_kernel<<<dim3(NH, L_SEQ / 64), blk, 0, stream>>>(Qh, Kh, VT, AOh);

  // output 1: AO @ wo -> fp32 [L, DIM]
  gemm_tn_f16<<<dim3(DIM / 64, L_SEQ / 64), blk, 0, stream>>>(AOh, woT, out, L_SEQ, DIM, QD);
}
